// DecoderLayer_72859825209415
// MI455X (gfx1250) — matrix operand order repaired and checked by reading, not compiled, not run on hardware
//
#include <hip/hip_runtime.h>

// ---------------------------------------------------------------------------
// Problem constants (match reference)
// ---------------------------------------------------------------------------
constexpr int Bb   = 4;
constexpr int Dd   = 1024;
constexpr int Ss   = 1024;
constexpr int SEe  = 1024;
constexpr int Hh   = 16;
constexpr int DFFf = 4096;
constexpr int DHh  = 64;   // Dd / Hh

typedef __attribute__((ext_vector_type(16))) __bf16 bf16x16;
typedef __attribute__((ext_vector_type(8)))  __bf16 bf16x8;
typedef __attribute__((ext_vector_type(8)))  float  floatx8;

// ---------------------------------------------------------------------------
// Helpers
// ---------------------------------------------------------------------------
__device__ inline __bf16 f2bf(float x) {
  unsigned u = __builtin_bit_cast(unsigned, x);
  u += 0x7FFFu + ((u >> 16) & 1u);          // round-to-nearest-even
  unsigned short h = (unsigned short)(u >> 16);
  return __builtin_bit_cast(__bf16, h);
}

__device__ inline void store_out(float* p, float v)  { *p = v; }
__device__ inline void store_out(__bf16* p, float v) { *p = f2bf(v); }

__device__ inline floatx8 wmma_bf16(bf16x16 a, bf16x16 b, floatx8 c) {
  return __builtin_amdgcn_wmma_f32_16x16x32_bf16(
      /*neg_a=*/false, a, /*neg_b=*/false, b,
      /*c_mod=*/(short)0, c, /*reuse_a=*/false, /*reuse_b=*/false);
}

__device__ inline bf16x16 combine(bf16x8 lo, bf16x8 hi) {
  return __builtin_shufflevector(lo, hi, 0, 1, 2, 3, 4, 5, 6, 7,
                                 8, 9, 10, 11, 12, 13, 14, 15);
}

// A fragment (16x32 bf16) from a row-major tile (LDS or global), leading dim
// `ld` elements.  ISA layout: lanes 0-15 row M=lane hold K={0..7,16..23};
// lanes 16-31 same rows hold K={8..15,24..31}.  All chunk offsets are 16B
// multiples -> two aligned b128 loads.
__device__ inline bf16x16 load_a_frag(const __bf16* base, int ld) {
  const int lane = threadIdx.x & 31;
  const int row  = lane & 15;
  const int kh   = lane >> 4;
  const __bf16* p = base + row * ld;
  bf16x8 lo = *reinterpret_cast<const bf16x8*>(p + kh * 8);
  bf16x8 hi = *reinterpret_cast<const bf16x8*>(p + 16 + kh * 8);
  return combine(lo, hi);
}

// B fragment (32x16 bf16) from a K-major tile: row = output column,
// 32 contiguous K values per row.  lanes 0-15: col N=lane, K={0..7,16..23};
// lanes 16-31: col N=lane-16, K={8..15,24..31}.  Two aligned b128 loads.
__device__ inline bf16x16 load_b_frag_kmajor(const __bf16* base, int ld) {
  const int lane = threadIdx.x & 31;
  const int col  = lane & 15;
  const int kh   = lane >> 4;
  const __bf16* p = base + col * ld + kh * 8;
  bf16x8 lo = *reinterpret_cast<const bf16x8*>(p);
  bf16x8 hi = *reinterpret_cast<const bf16x8*>(p + 16);
  return combine(lo, hi);
}

// ---------------------------------------------------------------------------
// GEMM: C[M,N] = act( A[M,K](bf16) x B[K,N](fp32 weights) + bias ).
// Block tile 128x128; 8 waves in a 2(M) x 4(N) grid, each wave 64x32
// (4x2 fragments, 8 WMMA per 32-wide K step).  Output fp32 or bf16.
// Next-tile global_prefetch hides HBM/L2 latency across the barrier.
// ---------------------------------------------------------------------------
template <typename TC, int HAS_BIAS, int RELU>
__global__ __launch_bounds__(256)
void gemm_a16_wmma(const __bf16* __restrict__ A, const float* __restrict__ Bm,
                   TC* __restrict__ C, const float* __restrict__ bias,
                   int M, int N, int K)
{
  __shared__ alignas(16) __bf16 As[128][40];    // row-major, K contiguous
  __shared__ alignas(16) __bf16 Bst[128][40];   // K-major: [n][k]

  const int tid  = threadIdx.x;
  const int lane = tid & 31;
  const int wid  = tid >> 5;
  const int wr   = wid & 1;        // wave row group (0..1) -> 64 rows
  const int wc   = wid >> 1;       // wave col group (0..3) -> 32 cols
  const int m0   = blockIdx.y * 128;
  const int n0   = blockIdx.x * 128;

  floatx8 zf = {};
  floatx8 acc[4][2];
#pragma unroll
  for (int i = 0; i < 4; ++i) { acc[i][0] = zf; acc[i][1] = zf; }

  for (int k0 = 0; k0 < K; k0 += 32) {
    // A tile: 128 rows x 32 K, bf16 -> pure 16-byte vector copies.
#pragma unroll
    for (int i = 0; i < 2; ++i) {
      int e8 = tid + i * 256;          // 512 chunks of 8 elements
      int r  = e8 >> 2;                // 4 chunks per row
      int c8 = (e8 & 3) * 8;
      *reinterpret_cast<bf16x8*>(&As[r][c8]) =
          *reinterpret_cast<const bf16x8*>(&A[(size_t)(m0 + r) * K + k0 + c8]);
    }
    // B tile: 32 K x 128 N fp32 weights, read coalesced along N,
    // converted and stored K-major.
#pragma unroll
    for (int i = 0; i < 16; ++i) {
      int e = tid + i * 256;           // 0..4095
      int k = e >> 7, n = e & 127;
      Bst[n][k] = f2bf(Bm[(size_t)(k0 + k) * N + n0 + n]);
    }
    // Prefetch next K-step tiles (speculative, cacheline granular).
    if (k0 + 32 < K) {
      if (tid < 128)   // A: 128 rows x 64B (one line per row)
        __builtin_prefetch(&A[(size_t)(m0 + tid) * K + k0 + 32], 0, 3);
      // B: 32 rows x 512B (8 lines per row) -> 256 addresses
      __builtin_prefetch(&Bm[(size_t)(k0 + 32 + (tid >> 3)) * N +
                             n0 + (tid & 7) * 16], 0, 3);
    }
    __syncthreads();

    bf16x16 bf0 = load_b_frag_kmajor(&Bst[wc * 32][0],      40);
    bf16x16 bf1 = load_b_frag_kmajor(&Bst[wc * 32 + 16][0], 40);
#pragma unroll
    for (int i = 0; i < 4; ++i) {
      bf16x16 a = load_a_frag(&As[wr * 64 + i * 16][0], 40);
      acc[i][0] = wmma_bf16(a, bf0, acc[i][0]);
      acc[i][1] = wmma_bf16(a, bf1, acc[i][1]);
    }
    __syncthreads();
  }

  const int col16 = lane & 15;
  const int mbase = (lane >> 4) * 8;
#pragma unroll
  for (int j = 0; j < 2; ++j) {
    const int col = n0 + wc * 32 + j * 16 + col16;
    float bv = HAS_BIAS ? bias[col] : 0.0f;
#pragma unroll
    for (int i = 0; i < 4; ++i) {
#pragma unroll
      for (int g = 0; g < 8; ++g) {
        int row = m0 + wr * 64 + i * 16 + mbase + g;
        float v = acc[i][j][g];
        if (HAS_BIAS) v += bv;
        if (RELU) v = fmaxf(v, 0.0f);
        store_out(&C[(size_t)row * N + col], v);
      }
    }
  }
}

// ---------------------------------------------------------------------------
// Fused flash-attention (WMMA QK^T + online softmax + WMMA PV), all-bf16 I/O.
// Q,K,V,O: [B, Sx, D] bf16, head h uses cols [h*64, h*64+64).
// grid = (Sq/128, H, B); block = 256 (8 waves, 16 query rows per wave).
// ---------------------------------------------------------------------------
__global__ __launch_bounds__(256)
void flash_attn_wmma(const __bf16* __restrict__ Q, const __bf16* __restrict__ Kg,
                     const __bf16* __restrict__ V, __bf16* __restrict__ O,
                     int Sq, int Sk, int causal)
{
  __shared__ alignas(16) __bf16 Ks[32][72];      // [key][feat] (feat contig)
  __shared__ alignas(16) __bf16 Vts[64][40];     // transposed: [feat][key]
  __shared__ alignas(16) __bf16 Ps[8][16][40];   // per-wave P tile

  const int tid  = threadIdx.x;
  const int lane = tid & 31;
  const int wid  = tid >> 5;
  const int b    = blockIdx.z;
  const int h    = blockIdx.y;
  const int q0   = blockIdx.x * 128 + wid * 16;

  const int coln = lane & 15;
  const int kh   = lane >> 4;
  const int mb   = kh * 8;

  // Q tile (16 x 64): two A fragments loaded straight from global as
  // aligned b128 pairs (no LDS staging, no conversion).
  bf16x16 qa[2];
  {
    const int row = lane & 15;
    const __bf16* qp = Q + ((size_t)b * Sq + q0 + row) * Dd + h * DHh;
#pragma unroll
    for (int c = 0; c < 2; ++c) {
      bf16x8 lo = *reinterpret_cast<const bf16x8*>(qp + c * 32 + kh * 8);
      bf16x8 hi = *reinterpret_cast<const bf16x8*>(qp + c * 32 + 16 + kh * 8);
      qa[c] = combine(lo, hi);
    }
  }

  floatx8 zf = {};
  floatx8 o[4];
  o[0] = zf; o[1] = zf; o[2] = zf; o[3] = zf;
  float mrow[8], lrow[8];
#pragma unroll
  for (int g = 0; g < 8; ++g) { mrow[g] = -3.0e38f; lrow[g] = 0.0f; }

  const float scale = 0.125f;  // 1/sqrt(DH=64)
  const int nkt = causal ? (blockIdx.x * 4 + 4) : (Sk / 32);

  for (int kt = 0; kt < nkt; ++kt) {
    const int kb = kt * 32;

    // Cooperative load: K tile [key][feat] via vector copy; V transposed
    // to [feat][key] (vector global load + 8 scalar LDS stores).
    {
      int r  = tid >> 3;            // 0..31 key
      int c8 = (tid & 7) * 8;       // feat chunk
      size_t gidx = ((size_t)b * Sk + kb + r) * Dd + h * DHh + c8;
      *reinterpret_cast<bf16x8*>(&Ks[r][c8]) =
          *reinterpret_cast<const bf16x8*>(&Kg[gidx]);
      bf16x8 vv = *reinterpret_cast<const bf16x8*>(&V[gidx]);
#pragma unroll
      for (int i = 0; i < 8; ++i) Vts[c8 + i][r] = vv[i];
      // Prefetch next key tile while this one is consumed.
      if (kt + 1 < nkt) {
        size_t gn = gidx + (size_t)32 * Dd;
        __builtin_prefetch(&Kg[gn], 0, 3);
        __builtin_prefetch(&V[gn], 0, 3);
      }
    }
    __syncthreads();

    // S(16x32) = Q(16x64) . K^T : two 16x16 C-frags, K split in 2 chunks.
    floatx8 s[2];
    s[0] = zf; s[1] = zf;
#pragma unroll
    for (int j = 0; j < 2; ++j) {
#pragma unroll
      for (int c = 0; c < 2; ++c) {
        bf16x16 bk = load_b_frag_kmajor(&Ks[j * 16][c * 32], 72);
        s[j] = wmma_bf16(qa[c], bk, s[j]);
      }
    }

    // Scale + causal mask + per-tile row max.
    float tmax[8];
#pragma unroll
    for (int g = 0; g < 8; ++g) {
      float s0 = s[0][g] * scale;
      float s1 = s[1][g] * scale;
      if (causal) {
        int qrow = q0 + mb + g;
        if (kb + coln      > qrow) s0 = -3.0e38f;
        if (kb + 16 + coln > qrow) s1 = -3.0e38f;
      }
      s[0][g] = s0; s[1][g] = s1;
      tmax[g] = fmaxf(s0, s1);
    }
#pragma unroll
    for (int off = 1; off < 16; off <<= 1) {
#pragma unroll
      for (int g = 0; g < 8; ++g)
        tmax[g] = fmaxf(tmax[g], __shfl_xor(tmax[g], off, 32));
    }

    // Online softmax update.
    float pr0[8], pr1[8], rsum[8];
#pragma unroll
    for (int g = 0; g < 8; ++g) {
      float mnew = fmaxf(mrow[g], tmax[g]);
      float corr = __expf(mrow[g] - mnew);
      mrow[g] = mnew;
      lrow[g] *= corr;
      o[0][g] *= corr; o[1][g] *= corr; o[2][g] *= corr; o[3][g] *= corr;
      float p0 = (s[0][g] <= -1.0e37f) ? 0.0f : __expf(s[0][g] - mnew);
      float p1 = (s[1][g] <= -1.0e37f) ? 0.0f : __expf(s[1][g] - mnew);
      pr0[g] = p0; pr1[g] = p1;
      rsum[g] = p0 + p1;
    }
#pragma unroll
    for (int off = 1; off < 16; off <<= 1) {
#pragma unroll
      for (int g = 0; g < 8; ++g)
        rsum[g] += __shfl_xor(rsum[g], off, 32);
    }
#pragma unroll
    for (int g = 0; g < 8; ++g) lrow[g] += rsum[g];

    // Re-layout P: C-frag -> per-wave LDS tile -> A-frag.
#pragma unroll
    for (int g = 0; g < 8; ++g) {
      int r = mb + g;
      Ps[wid][r][coln]      = f2bf(pr0[g]);
      Ps[wid][r][16 + coln] = f2bf(pr1[g]);
    }
    bf16x16 pa = load_a_frag(&Ps[wid][0][0], 40);

    // O(16x64) += P(16x32) . V(32x64): B fragment column = dh (row of Vts),
    // K dim = key (contiguous in the transposed tile).
#pragma unroll
    for (int j = 0; j < 4; ++j) {
      bf16x16 bv = load_b_frag_kmajor(&Vts[j * 16][0], 40);
      o[j] = wmma_bf16(pa, bv, o[j]);
    }
    __syncthreads();
  }

  // Normalize and write O (bf16).
  float linv[8];
#pragma unroll
  for (int g = 0; g < 8; ++g) linv[g] = 1.0f / (lrow[g] + 1.0e-20f);
#pragma unroll
  for (int j = 0; j < 4; ++j) {
#pragma unroll
    for (int g = 0; g < 8; ++g) {
      int row = q0 + mb + g;
      O[((size_t)b * Sq + row) * Dd + h * DHh + j * 16 + coln] =
          f2bf(o[j][g] * linv[g]);
    }
  }
}

// ---------------------------------------------------------------------------
// Transpose [B, D, S] (fp32) -> [B, S, D] (bf16)
// ---------------------------------------------------------------------------
__global__ __launch_bounds__(256)
void transpose_bds_bf(const float* __restrict__ in, __bf16* __restrict__ out,
                      int Dn, int Sn)
{
  __shared__ float t[32][33];
  const int b  = blockIdx.z;
  const int d0 = blockIdx.y * 32;
  const int s0 = blockIdx.x * 32;
  const int tx = threadIdx.x & 31;
  const int ty = threadIdx.x >> 5;   // 0..7
#pragma unroll
  for (int i = 0; i < 32; i += 8)
    t[ty + i][tx] = in[((size_t)b * Dn + d0 + ty + i) * Sn + s0 + tx];
  __syncthreads();
#pragma unroll
  for (int i = 0; i < 32; i += 8)
    out[((size_t)b * Sn + s0 + ty + i) * Dn + d0 + tx] = f2bf(t[tx][ty + i]);
}

// ---------------------------------------------------------------------------
// Residual + custom global norm:  y = a[b,d,s] + cT[b,s,d]
//   mean over whole [D,S] slab, divide by full L2 norm of centered slab.
// red[b] = sum, red[B+b] = sumsq.  3 passes; y staged in dst (fp32).
// Optional transposed bf16 output for the next matrix stage.
// ---------------------------------------------------------------------------
__global__ void zero_red(float* red, int n) {
  int i = blockIdx.x * blockDim.x + threadIdx.x;
  if (i < n) red[i] = 0.0f;
}

__global__ __launch_bounds__(256)
void rn_pass1(const float* __restrict__ a, const float* __restrict__ cT,
              float* __restrict__ y, float* __restrict__ red)
{
  __shared__ float sm[256];
  const int b = blockIdx.y;
  const size_t base = (size_t)b * Dd * Ss;
  float local = 0.0f;
#pragma unroll
  for (int i = 0; i < 4; ++i) {
    int j = blockIdx.x * 1024 + i * 256 + threadIdx.x;
    int d = j >> 10;           // / Ss (=1024)
    int s = j & 1023;
    float v = a[base + j] + cT[(size_t)b * Ss * Dd + (size_t)s * Dd + d];
    y[base + j] = v;
    local += v;
  }
  sm[threadIdx.x] = local;
  __syncthreads();
  for (int w = 128; w > 0; w >>= 1) {
    if (threadIdx.x < w) sm[threadIdx.x] += sm[threadIdx.x + w];
    __syncthreads();
  }
  if (threadIdx.x == 0) atomicAdd(&red[b], sm[0]);
}

__global__ __launch_bounds__(256)
void rn_pass2(const float* __restrict__ y, float* __restrict__ red)
{
  __shared__ float sm[256];
  const int b = blockIdx.y;
  const size_t base = (size_t)b * Dd * Ss;
  const float mean = red[b] * (1.0f / ((float)Dd * (float)Ss));
  float local = 0.0f;
#pragma unroll
  for (int i = 0; i < 4; ++i) {
    int j = blockIdx.x * 1024 + i * 256 + threadIdx.x;
    float v = y[base + j] - mean;
    local += v * v;
  }
  sm[threadIdx.x] = local;
  __syncthreads();
  for (int w = 128; w > 0; w >>= 1) {
    if (threadIdx.x < w) sm[threadIdx.x] += sm[threadIdx.x + w];
    __syncthreads();
  }
  if (threadIdx.x == 0) atomicAdd(&red[Bb + b], sm[0]);
}

__global__ __launch_bounds__(256)
void rn_pass3(float* __restrict__ y, const float* __restrict__ red,
              __bf16* __restrict__ outT /* [B,S,D] bf16 or nullptr */)
{
  const int b = blockIdx.y;
  const size_t base = (size_t)b * Dd * Ss;
  const float mean = red[b] * (1.0f / ((float)Dd * (float)Ss));
  const float rstd = 1.0f / (sqrtf(red[Bb + b]) + 1.0e-7f);
#pragma unroll
  for (int i = 0; i < 4; ++i) {
    int j = blockIdx.x * 1024 + i * 256 + threadIdx.x;
    int d = j >> 10;
    int s = j & 1023;
    float v = (y[base + j] - mean) * rstd;
    y[base + j] = v;
    if (outT) outT[(size_t)b * Ss * Dd + (size_t)s * Dd + d] = f2bf(v);
  }
}

// ---------------------------------------------------------------------------
// Host-side launch orchestration
// ---------------------------------------------------------------------------
template <typename TC>
static inline void launch_gemm(const __bf16* A, const float* Bm, TC* C,
                               const float* bias, int M, int N, int K,
                               int relu, hipStream_t stream) {
  dim3 g(N / 128, M / 128), blk(256);
  if (bias && relu)
    gemm_a16_wmma<TC, 1, 1><<<g, blk, 0, stream>>>(A, Bm, C, bias, M, N, K);
  else if (bias)
    gemm_a16_wmma<TC, 1, 0><<<g, blk, 0, stream>>>(A, Bm, C, bias, M, N, K);
  else
    gemm_a16_wmma<TC, 0, 0><<<g, blk, 0, stream>>>(A, Bm, C, bias, M, N, K);
}

static inline void launch_norm(const float* a, const float* cT, float* dst,
                               __bf16* dstT, float* red, hipStream_t stream) {
  zero_red<<<1, 64, 0, stream>>>(red, 2 * Bb);
  dim3 g(Dd * Ss / 1024, Bb), blk(256);
  rn_pass1<<<g, blk, 0, stream>>>(a, cT, dst, red);
  rn_pass2<<<g, blk, 0, stream>>>(dst, red);
  rn_pass3<<<g, blk, 0, stream>>>(dst, red, dstT);
}

extern "C" void kernel_launch(void* const* d_in, const int* in_sizes, int n_in,
                              void* d_out, int out_size, void* d_ws, size_t ws_size,
                              hipStream_t stream) {
  const float* embedding = (const float*)d_in[0];   // [B, D, SE]
  const float* other     = (const float*)d_in[1];   // [B, D, S]
  const float* Wq_s = (const float*)d_in[2];
  const float* Wk_s = (const float*)d_in[3];
  const float* Wv_s = (const float*)d_in[4];
  const float* Wo_s = (const float*)d_in[5];
  const float* Wq_c = (const float*)d_in[6];
  const float* Wk_c = (const float*)d_in[7];
  const float* Wv_c = (const float*)d_in[8];
  const float* Wo_c = (const float*)d_in[9];
  const float* W1   = (const float*)d_in[10];
  const float* b1   = (const float*)d_in[11];
  const float* W2   = (const float*)d_in[12];
  const float* b2   = (const float*)d_in[13];
  float* out = (float*)d_out;                       // [B, D, S]

  const size_t T = (size_t)Bb * Ss * Dd;            // 4M elements
  char* wsb = (char*)d_ws;
  // bf16 operand buffers
  __bf16* bT0 = (__bf16*)(wsb + 0 * T);        // xT -> paT -> peT  [B,S,D]
  __bf16* bQ  = (__bf16*)(wsb + 2 * T);
  __bf16* bK  = (__bf16*)(wsb + 4 * T);
  __bf16* bV  = (__bf16*)(wsb + 6 * T);
  __bf16* bX  = (__bf16*)(wsb + 8 * T);        // attn out / embT (sequential)
  __bf16* bFF = (__bf16*)(wsb + 10 * T);       // MLP hidden [B,S,DFF] bf16
  // fp32 buffers
  float*  fSA = (float*)(wsb + 18 * T);        // SA -> CA -> ff2   [B,S,D]
  float*  f6  = (float*)(wsb + 22 * T);        // post_attention    [B,D,S]
  float*  f7  = (float*)(wsb + 26 * T);        // post_embedding    [B,D,S]
  float*  red = (float*)(wsb + 30 * T);        // 2*B reduction scalars

  const int M = Bb * Ss;    // 4096 rows for all GEMMs
  dim3 tgrid(Ss / 32, Dd / 32, Bb), tblk(256);
  dim3 agrid(Ss / 128, Hh, Bb), ablk(256);

  // ---- Self attention -----------------------------------------------------
  transpose_bds_bf<<<tgrid, tblk, 0, stream>>>(other, bT0, Dd, Ss);   // xT
  launch_gemm(bT0, Wq_s, bQ, (const float*)nullptr, M, Dd, Dd, 0, stream);
  launch_gemm(bT0, Wk_s, bK, (const float*)nullptr, M, Dd, Dd, 0, stream);
  launch_gemm(bT0, Wv_s, bV, (const float*)nullptr, M, Dd, Dd, 0, stream);
  flash_attn_wmma<<<agrid, ablk, 0, stream>>>(bQ, bK, bV, bX, Ss, Ss, 1);
  launch_gemm(bX, Wo_s, fSA, (const float*)nullptr, M, Dd, Dd, 0, stream);
  // post_attention = norm(other + SA^T); also emit bf16 paT into bT0
  launch_norm(other, fSA, f6, bT0, red, stream);

  // ---- Cross attention ----------------------------------------------------
  transpose_bds_bf<<<tgrid, tblk, 0, stream>>>(embedding, bX, Dd, SEe); // embT
  launch_gemm(bT0, Wq_c, bQ, (const float*)nullptr, M, Dd, Dd, 0, stream);
  launch_gemm(bX, Wk_c, bK, (const float*)nullptr, M, Dd, Dd, 0, stream);
  launch_gemm(bX, Wv_c, bV, (const float*)nullptr, M, Dd, Dd, 0, stream);
  flash_attn_wmma<<<agrid, ablk, 0, stream>>>(bQ, bK, bV, bX, Ss, SEe, 0);
  launch_gemm(bX, Wo_c, fSA, (const float*)nullptr, M, Dd, Dd, 0, stream);
  // post_embedding = norm(post_attention + CA^T); emit bf16 peT into bT0
  launch_norm(f6, fSA, f7, bT0, red, stream);

  // ---- MLP ----------------------------------------------------------------
  launch_gemm(bT0, W1, bFF, b1, M, DFFf, Dd, 1, stream);   // relu(peT@W1+b1)
  launch_gemm(bFF, W2, fSA, b2, M, Dd, DFFf, 0, stream);   // @W2+b2 (fp32)
  // out = norm(post_embedding + ff^T)
  launch_norm(f7, fSA, out, (__bf16*)nullptr, red, stream);
}
